// Head_81423989997812
// MI455X (gfx1250) — matrix operand order repaired and checked by reading, not compiled, not run on hardware
//
#include <hip/hip_runtime.h>
#include <hip/hip_bf16.h>

// ---------------------------------------------------------------------------
// Single-head causal attention for MI455X (gfx1250, wave32, WMMA bf16 path).
//   B=8, S=2048, E=1024, H=64.  M = B*S = 16384 rows.
// Kernel 0: convert W{q,k,v} -> bf16 transposed [w][h][e] (contiguous B frags).
// Kernel 1: q/k/v projections via v_wmma_f32_16x16x32_bf16; copy-free
//           ping-pong double buffering (k-loop unrolled x2).
// Kernel 2: flash attention, wave-per-16-query-rows, 32-key blocks, online
//           softmax in registers; next K block + V fragments loaded under
//           the softmax VALU stretch; P relaid via per-wave LDS.
// ---------------------------------------------------------------------------

typedef __attribute__((ext_vector_type(16))) __bf16 v16bf;
typedef __attribute__((ext_vector_type(8)))  __bf16 v8bf;
typedef __attribute__((ext_vector_type(8)))  float  v8f;

#define EMBED 1024
#define HEAD  64
#define SEQ   2048
#define BATCH 8
#define MROWS (BATCH * SEQ)   // 16384

static __device__ __forceinline__ v8f wmma_bf16(v16bf a, v16bf b, v8f c) {
  // (neg_a, A, neg_b, B, c_mod, C, reuse_a, reuse_b)
  return __builtin_amdgcn_wmma_f32_16x16x32_bf16(false, a, false, b,
                                                 (short)0, c, false, false);
}

static __device__ __forceinline__ v16bf join16(v8bf lo, v8bf hi) {
  return __builtin_shufflevector(lo, hi, 0, 1, 2, 3, 4, 5, 6, 7,
                                         8, 9, 10, 11, 12, 13, 14, 15);
}

// ---------------------------------------------------------------------------
// Kernel 0: W -> bf16 transposed.  wt[w][h][e] = W_w[e*HEAD + h].
// ---------------------------------------------------------------------------
__global__ void __launch_bounds__(256)
wcvt_kernel(const float* __restrict__ Wk, const float* __restrict__ Wq,
            const float* __restrict__ Wv, __bf16* __restrict__ wt) {
  const int idx = blockIdx.x * 256 + threadIdx.x;
  const int w   = idx >> 16;            // 65536 elements per matrix
  const int rem = idx & 65535;
  const int h   = rem >> 10;
  const int e   = rem & 1023;
  const float* __restrict__ W = (w == 0) ? Wq : (w == 1) ? Wk : Wv;
  wt[idx] = (__bf16)W[e * HEAD + h];
}

// ---------------------------------------------------------------------------
// Kernel 1: QKV projection.  grid = 1024 (16-row M tiles), block = 96 (3 waves)
// wave 0 -> q, wave 1 -> k, wave 2 -> v (transposed store).
// ---------------------------------------------------------------------------
__global__ void __launch_bounds__(96)
qkv_proj_kernel(const float* __restrict__ x,
                const __bf16* __restrict__ wt,  // [3][HEAD][EMBED] bf16
                __bf16* __restrict__ qb,        // [MROWS, HEAD] row-major
                __bf16* __restrict__ kb,        // [MROWS, HEAD] row-major
                __bf16* __restrict__ vT)        // [HEAD, MROWS] (transposed)
{
  const int lane  = threadIdx.x & 31;
  const int wave  = threadIdx.x >> 5;          // 0=q, 1=k, 2=v
  const int tile  = blockIdx.x;                // 16-row tile of M
  const int m15   = lane & 15;
  const int hi    = lane >> 4;                 // lane half-group
  const int abase = hi ? 8 : 0;                // unused; A frags also use kbase
  const int kbase = hi ? 16 : 0;               // B-frag K offset for this group
  const int row   = tile * 16 + m15;

  const __bf16* __restrict__ Wt = wt + (size_t)wave * HEAD * EMBED;
  const float*  __restrict__ xrow = x + (size_t)row * EMBED;

  auto loadA = [&](int ko) -> v16bf {
    float xf[16];
    const float* xr = xrow + ko;
    *(float4*)(&xf[0])  = *(const float4*)(xr + kbase);
    *(float4*)(&xf[4])  = *(const float4*)(xr + kbase + 4);
    *(float4*)(&xf[8])  = *(const float4*)(xr + kbase + 8);
    *(float4*)(&xf[12]) = *(const float4*)(xr + kbase + 12);
    v16bf a;
#pragma unroll
    for (int e = 0; e < 16; ++e) a[e] = (__bf16)xf[e];
    return a;
  };
  auto loadB = [&](int ko, int nt) -> v16bf {
    return *(const v16bf*)(Wt + (size_t)(nt * 16 + m15) * EMBED + ko + kbase);
  };

  v8f acc[4] = {};                             // 16x64 f32 accumulator tile

  // Copy-free ping-pong: two register stages, k-loop advances by 64.
  v16bf aA, bA[4], aB, bB[4];
  aA = loadA(0);
#pragma unroll
  for (int nt = 0; nt < 4; ++nt) bA[nt] = loadB(0, nt);

  for (int ko = 0; ko < EMBED; ko += 64) {
    // Stage B loads (ko+32) issued before stage-A WMMAs.
    aB = loadA(ko + 32);
#pragma unroll
    for (int nt = 0; nt < 4; ++nt) bB[nt] = loadB(ko + 32, nt);

#pragma unroll
    for (int nt = 0; nt < 4; ++nt) acc[nt] = wmma_bf16(aA, bA[nt], acc[nt]);

    // Stage A loads (ko+64) issued before stage-B WMMAs.
    if (ko + 64 < EMBED) {
      aA = loadA(ko + 64);
#pragma unroll
      for (int nt = 0; nt < 4; ++nt) bA[nt] = loadB(ko + 64, nt);
    }

#pragma unroll
    for (int nt = 0; nt < 4; ++nt) acc[nt] = wmma_bf16(aB, bB[nt], acc[nt]);
  }

  // Store bf16 results.  C/D layout: vgpr r, lane l -> M = r + 8*(l>>4),
  // N = l & 15 (+ 16*nt).  Wave branch hoisted out of the loops.
  if (wave == 0) {
#pragma unroll
    for (int nt = 0; nt < 4; ++nt)
#pragma unroll
      for (int r = 0; r < 8; ++r)
        qb[(size_t)(tile * 16 + r + 8 * hi) * HEAD + nt * 16 + m15] =
            (__bf16)acc[nt][r];
  } else if (wave == 1) {
#pragma unroll
    for (int nt = 0; nt < 4; ++nt)
#pragma unroll
      for (int r = 0; r < 8; ++r)
        kb[(size_t)(tile * 16 + r + 8 * hi) * HEAD + nt * 16 + m15] =
            (__bf16)acc[nt][r];
  } else {
#pragma unroll
    for (int nt = 0; nt < 4; ++nt)
#pragma unroll
      for (int r = 0; r < 8; ++r)
        vT[(size_t)(nt * 16 + m15) * MROWS + tile * 16 + r + 8 * hi] =
            (__bf16)acc[nt][r];
  }
}

// ---------------------------------------------------------------------------
// Kernel 2: flash attention.  grid = 256, block = 128 (4 waves).
// Each wave owns one 16-query-row tile: t = blockIdx*4 + wave  (1024 tiles).
// ---------------------------------------------------------------------------
__global__ void __launch_bounds__(128)
flash_attn_kernel(const __bf16* __restrict__ qb,
                  const __bf16* __restrict__ kb,
                  const __bf16* __restrict__ vT,
                  float* __restrict__ out)
{
  __shared__ __align__(32) __bf16 lds_p[4][16 * 32]; // per-wave P (16x32)

  const int lane  = threadIdx.x & 31;
  const int wave  = threadIdx.x >> 5;
  const int t     = blockIdx.x * 4 + wave;     // global q-tile id, < 1024
  const int b     = t >> 7;                    // batch (128 tiles per batch)
  const int qt    = t & 127;                   // q tile within batch
  const int m15   = lane & 15;
  const int hi    = lane >> 4;
  const int abase = hi ? 8 : 0;
  const int kbase = hi ? 16 : 0;
  const size_t rowbase = (size_t)b * SEQ;

  // --- Q A-fragments for the two K=32 chunks of H=64 (b128 loads) ---
  v16bf aq[2];
  {
    const __bf16* qr = qb + (rowbase + qt * 16 + m15) * HEAD;
#pragma unroll
    for (int s = 0; s < 2; ++s) {
      const v8bf lo = *(const v8bf*)(qr + s * 32 + kbase);
      const v8bf hh = *(const v8bf*)(qr + s * 32 + kbase + 8);
      aq[s] = join16(lo, hh);
    }
  }

  // K B-fragment loader: bk[s*2+h0] for score subtile s, K-chunk h0.
  auto loadBK = [&](int key0, v16bf* bk) {
#pragma unroll
    for (int s = 0; s < 2; ++s)
#pragma unroll
      for (int h0 = 0; h0 < 2; ++h0)
        bk[s * 2 + h0] =
            *(const v16bf*)(kb + (rowbase + key0 + s * 16 + m15) * HEAD +
                            h0 * 32 + kbase);
  };

  v8f   acc[4] = {};                           // 16x64 output accumulator
  float mrow[8], lrow[8];
#pragma unroll
  for (int r = 0; r < 8; ++r) { mrow[r] = -3.0e38f; lrow[r] = 0.0f; }

  const float sc = 0.03125f;                   // 1024^-0.5
  const int nblk = (qt * 16 + 15) / 32 + 1;    // 32-key blocks needed (causal)

  v16bf bkC[4];
  loadBK(0, bkC);

  for (int j = 0; j < nblk; ++j) {
    const int key0 = j * 32;

    // --- scores: two 16x16 subtiles, each = Q(16x64) . K^T chunk ---
    v8f S[2];
#pragma unroll
    for (int s = 0; s < 2; ++s) {
      v8f c = {};
      c = wmma_bf16(aq[0], bkC[s * 2 + 0], c);
      c = wmma_bf16(aq[1], bkC[s * 2 + 1], c);
      S[s] = c;
    }

    // --- issue V fragments + next K block; both hide under the softmax ---
    v16bf bv[4];
#pragma unroll
    for (int nt = 0; nt < 4; ++nt)
      bv[nt] = *(const v16bf*)(vT + (size_t)(nt * 16 + m15) * MROWS +
                               rowbase + key0 + kbase);
    v16bf bkN[4];
    if (j + 1 < nblk) loadBK(key0 + 32, bkN);
    if (j + 2 < nblk)   // prefetch j+2 (global_prefetch_b8)
      __builtin_prefetch(kb + (rowbase + key0 + 64 + m15) * HEAD, 0, 1);

    // --- scale + causal mask (pure VALU selects; EXEC untouched) ---
#pragma unroll
    for (int s = 0; s < 2; ++s) {
#pragma unroll
      for (int r = 0; r < 8; ++r) {
        const int rg = qt * 16 + r + 8 * hi;
        const int cg = key0 + s * 16 + m15;
        const float v = S[s][r] * sc;
        S[s][r] = (cg > rg) ? -3.0e38f : v;
      }
    }

    // --- online softmax (row = 16 lanes of one half-wave, per C layout) ---
#pragma unroll
    for (int r = 0; r < 8; ++r) {
      float mx = fmaxf(S[0][r], S[1][r]);
#pragma unroll
      for (int off = 8; off >= 1; off >>= 1)
        mx = fmaxf(mx, __shfl_xor(mx, off, 32));
      const float mnew = fmaxf(mrow[r], mx);
      const float corr = __expf(mrow[r] - mnew);
      const float p0 = __expf(S[0][r] - mnew);
      const float p1 = __expf(S[1][r] - mnew);
      float ps = p0 + p1;
#pragma unroll
      for (int off = 8; off >= 1; off >>= 1)
        ps += __shfl_xor(ps, off, 32);
      lrow[r] = lrow[r] * corr + ps;
      mrow[r] = mnew;
      S[0][r] = p0;
      S[1][r] = p1;
#pragma unroll
      for (int nt = 0; nt < 4; ++nt) acc[nt][r] *= corr;
    }

    // --- relayout P (C layout -> A layout) via per-wave LDS ---
    __bf16* pl = lds_p[wave];
#pragma unroll
    for (int s = 0; s < 2; ++s)
#pragma unroll
      for (int r = 0; r < 8; ++r)
        pl[(r + 8 * hi) * 32 + s * 16 + m15] = (__bf16)S[s][r];
    // Same-wave DS ops complete in order; no barrier needed (private region).
    v16bf ap;
    {
      const __bf16* pr = pl + m15 * 32;
      const v8bf lo = *(const v8bf*)(pr + kbase);
      const v8bf hh = *(const v8bf*)(pr + kbase + 8);
      ap = join16(lo, hh);
    }

    // --- acc += P(16x32) . V(32x64) ---
#pragma unroll
    for (int nt = 0; nt < 4; ++nt) acc[nt] = wmma_bf16(ap, bv[nt], acc[nt]);

#pragma unroll
    for (int i = 0; i < 4; ++i) bkC[i] = bkN[i];
  }

  // --- normalize and store f32 output ---
#pragma unroll
  for (int nt = 0; nt < 4; ++nt) {
#pragma unroll
    for (int r = 0; r < 8; ++r) {
      const float o = acc[nt][r] / lrow[r];
      out[(rowbase + qt * 16 + r + 8 * hi) * HEAD + nt * 16 + m15] = o;
    }
  }
}

// ---------------------------------------------------------------------------
extern "C" void kernel_launch(void* const* d_in, const int* in_sizes, int n_in,
                              void* d_out, int out_size, void* d_ws,
                              size_t ws_size, hipStream_t stream) {
  const float* x  = (const float*)d_in[0];
  const float* Wk = (const float*)d_in[1];
  const float* Wq = (const float*)d_in[2];
  const float* Wv = (const float*)d_in[3];

  // Workspace: q, k (row-major), vT (transposed), WT (bf16 weights).  ~6.4 MB.
  __bf16* qb = (__bf16*)d_ws;
  __bf16* kb = qb + (size_t)MROWS * HEAD;
  __bf16* vT = kb + (size_t)MROWS * HEAD;
  __bf16* wt = vT + (size_t)MROWS * HEAD;      // 3*HEAD*EMBED bf16

  wcvt_kernel<<<768, 256, 0, stream>>>(Wk, Wq, Wv, wt);
  qkv_proj_kernel<<<MROWS / 16, 96, 0, stream>>>(x, wt, qb, kb, vT);
  flash_attn_kernel<<<(MROWS / 16) / 4, 128, 0, stream>>>(qb, kb, vT,
                                                          (float*)d_out);
}
